// DefocusBlur_32040456028522
// MI455X (gfx1250) — matrix operand order repaired and checked by reading, not compiled, not run on hardware
//
#include <hip/hip_runtime.h>
#include <hip/hip_bf16.h>

typedef __attribute__((ext_vector_type(16))) _Float16 v16h;
typedef __attribute__((ext_vector_type(8)))  _Float16 v8h;
typedef __attribute__((ext_vector_type(8)))  float    v8f;

#define TILE_W 64
#define TILE_H 64
#define STG_W  80     // 64 + 2*8 halo
#define STG_H  80     // 64 + 2*8 halo
#define SA     88     // staged row stride (halves): 176B, 16B-aligned, bank-padded
#define TBS    24     // weight-table per-lane stride (halves): 48B, 16B-aligned

union V16 { v16h v; v8h h[2]; };

// D = A x B + C ; A = plain 16x32 f16 input tile (M=out row, K=input col),
// B = banded Toeplitz weights per kernel row i, C chains the 17-row vertical sum.
__global__ __launch_bounds__(128)
void defocus17_wmma(const float* __restrict__ img, float* __restrict__ out, int planes)
{
    __shared__ float  disk[17 * 17];
    __shared__ float  wf[17 * 17];
    __shared__ _Float16 __align__(16) sHi[STG_H * SA];
    __shared__ _Float16 __align__(16) sLo[STG_H * SA];
    __shared__ _Float16 __align__(16) TB[17 * 32 * TBS];   // B in WMMA lane layout

    const int tid   = threadIdx.x;
    const int X0    = blockIdx.x * TILE_W;
    const int Y0    = blockIdx.y * TILE_H;
    const int plane = blockIdx.z;
    const size_t planeOff = (size_t)plane * 512 * 512;

    // ---- 1) aliased disk, radius 8, 17x17; 197 lattice points inside r^2<=64
    for (int idx = tid; idx < 289; idx += 128) {
        int dy = idx / 17 - 8, dx = idx % 17 - 8;
        disk[idx] = (dx * dx + dy * dy <= 64) ? (1.0f / 197.0f) : 0.0f;
    }
    __syncthreads();
    // ---- 2) 3x3 Gaussian (sigma=0.5) with reflect-101 border -> wf[17][17]
    for (int idx = tid; idx < 289; idx += 128) {
        int i = idx / 17, j = idx % 17;
        float g1 = expf(-2.0f);                 // exp(-1/(2*0.5^2))
        float gs = 1.0f + 2.0f * g1;
        float g[3] = { g1 / gs, 1.0f / gs, g1 / gs };
        float acc = 0.0f;
        #pragma unroll
        for (int a = -1; a <= 1; ++a)
            #pragma unroll
            for (int b = -1; b <= 1; ++b) {
                int ii = i + a; ii = ii < 0 ? -ii : (ii > 16 ? 32 - ii : ii);
                int jj = j + b; jj = jj < 0 ? -jj : (jj > 16 ? 32 - jj : jj);
                acc += g[a + 1] * g[b + 1] * disk[ii * 17 + jj];
            }
        wf[idx] = acc;
    }
    __syncthreads();
    // ---- 3) B tables, pre-swizzled into WMMA B layout: lane L (n=L&15,h=L>>4),
    //         element s holds K = (s<8 ? 8*h+s : 16+8*h+(s-8)) ; B[k][n] = w[i][k-n] banded.
    for (int idx = tid; idx < 17 * 32 * 16; idx += 128) {
        int s = idx & 15;
        int L = (idx >> 4) & 31;
        int i = idx >> 9;
        int n = L & 15, h = L >> 4;
        int j = (s < 8 ? 8 * h + s : 8 + 8 * h + s) - n;   // k - n
        float v = (j >= 0 && j <= 16) ? wf[i * 17 + j] : 0.0f;
        TB[(i * 32 + L) * TBS + s] = (_Float16)v;
    }
    // ---- 4) stage input with reflect-101 halo, split f32 -> f16 hi + f16 lo
    for (int idx = tid; idx < STG_H * STG_W; idx += 128) {
        int rr = idx / STG_W, cc = idx % STG_W;
        int gr = Y0 - 8 + rr; gr = gr < 0 ? -gr : (gr > 511 ? 1022 - gr : gr);
        int gc = X0 - 8 + cc; gc = gc < 0 ? -gc : (gc > 511 ? 1022 - gc : gc);
        float v = img[planeOff + (size_t)gr * 512 + gc];
        _Float16 hi = (_Float16)v;
        sHi[rr * SA + cc] = hi;
        sLo[rr * SA + cc] = (_Float16)(v - (float)hi);
    }
    __syncthreads();

    // ---- 5) compute: wave w owns output rows [Y0+16w, +16), cols [X0, X0+64)
    const int wv   = tid >> 5;      // wave32
    const int lane = tid & 31;
    const int n    = lane & 15, h = lane >> 4;
    const int Yw   = Y0 + 16 * wv;

    v8f dH[4], dL[4];
    #pragma unroll
    for (int t = 0; t < 4; ++t) { dH[t] = (v8f)0.0f; dL[t] = (v8f)0.0f; }

    for (int i = 0; i < 17; ++i) {
        V16 B;
        const _Float16* tb = &TB[(i * 32 + lane) * TBS];
        B.h[0] = *(const v8h*)(tb);
        B.h[1] = *(const v8h*)(tb + 8);

        // A lane layout: elems 0..7 -> K = 8h+e ; elems 8..15 -> K = 16+8h+(e-8)
        const int rs = 16 * wv + n + i;          // staged input row (origin Y0-8)
        const _Float16* rH = &sHi[rs * SA];
        const _Float16* rL = &sLo[rs * SA];

        V16 aH, aL;
        aH.h[0] = *(const v8h*)(rH + 8 * h);
        aH.h[1] = *(const v8h*)(rH + 16 + 8 * h);
        aL.h[0] = *(const v8h*)(rL + 8 * h);
        aL.h[1] = *(const v8h*)(rL + 16 + 8 * h);

        #pragma unroll
        for (int t = 0; t < 4; ++t) {
            dH[t] = __builtin_amdgcn_wmma_f32_16x16x32_f16(
                        false, aH.v, false, B.v, (short)0, dH[t], false, false);
            dL[t] = __builtin_amdgcn_wmma_f32_16x16x32_f16(
                        false, aL.v, false, B.v, (short)0, dL[t], false, false);
            if (t < 3) {        // slide window 16 cols: K(0..15) <- old K(16..31)
                aH.h[0] = aH.h[1];
                aH.h[1] = *(const v8h*)(rH + 16 * (t + 1) + 16 + 8 * h);
                aL.h[0] = aL.h[1];
                aL.h[1] = *(const v8h*)(rL + 16 * (t + 1) + 16 + 8 * h);
            }
        }
    }

    // ---- 6) store: C layout -> lane n = column, VGPR j = row (j+8h); coalesced
    #pragma unroll
    for (int t = 0; t < 4; ++t) {
        v8f r = dH[t] + dL[t];
        #pragma unroll
        for (int j = 0; j < 8; ++j) {
            int row = Yw + j + 8 * h;
            out[planeOff + (size_t)row * 512 + X0 + 16 * t + n] = r[j];
        }
    }
}

extern "C" void kernel_launch(void* const* d_in, const int* in_sizes, int n_in,
                              void* d_out, int out_size, void* d_ws, size_t ws_size,
                              hipStream_t stream)
{
    const float* img = (const float*)d_in[0];
    float* o = (float*)d_out;
    const int planes = in_sizes[0] / (512 * 512);   // 32*3 = 96
    dim3 grid(512 / TILE_W, 512 / TILE_H, planes);
    dim3 block(128);
    defocus17_wmma<<<grid, block, 0, stream>>>(img, o, planes);
}
